// MaskedMultiHeadAttention_83597243449800
// MI455X (gfx1250) — compile-verified, matrix operand order checked by reading, not run on hardware
//
#include <hip/hip_runtime.h>

typedef unsigned short u16;
typedef __attribute__((ext_vector_type(16))) __bf16 v16bf;
typedef __attribute__((ext_vector_type(8)))  float  v8f;
typedef __attribute__((ext_vector_type(4)))  unsigned int u32x4;
typedef __attribute__((ext_vector_type(8)))  int i32x8;
typedef __attribute__((ext_vector_type(4)))  int i32x4;

#define D_MODEL  1024
#define N_BATCH  4
#define SEQ      2048
#define N_HEADS  16
#define HEAD_DIM 64
#define SCALE    0.03125f   // D_MODEL^-0.5 = 1/32 (reference scales by full model dim)

// ---- feature detection -------------------------------------------------------
#if defined(__has_builtin)
#  if __has_builtin(__builtin_amdgcn_tensor_load_to_lds)
#    define USE_TDM 1
#  endif
#endif
#ifndef USE_TDM
#  define USE_TDM 0
#endif

#if defined(__has_builtin) && __has_builtin(__builtin_amdgcn_s_wait_tensorcnt)
#  define WAIT_TENSORCNT(n) __builtin_amdgcn_s_wait_tensorcnt(n)
#else
#  define WAIT_TENSORCNT(n) asm volatile("s_wait_tensorcnt %0" :: "i"(n) : "memory")
#endif

__device__ __forceinline__ u16 f2bf(float f) {
    unsigned u = __float_as_uint(f);
    u += 0x7FFFu + ((u >> 16) & 1u);       // round-to-nearest-even-ish
    return (u16)(u >> 16);
}

union FragU { uint4 q[2]; v16bf v; };

// Per-lane bf16 WMMA fragment load (A 16x32 or B 32x16), per ISA 16-bit layout:
//   elements 0..7  = p[kb + 0 .. kb + 7]
//   elements 8..15 = p[kb + 16 .. kb + 23]     kb = (lane>=16) ? 8 : 0
__device__ __forceinline__ v16bf load_frag(const u16* p, int kb) {
    FragU u;
    u.q[0] = *(const uint4*)(p + kb);
    u.q[1] = *(const uint4*)(p + kb + 16);
    return u.v;
}

__device__ __forceinline__ v8f wmma_bf16(v16bf a, v16bf b, v8f c) {
    return __builtin_amdgcn_wmma_f32_16x16x32_bf16(false, a, false, b, (short)0, c, false, false);
}

#if USE_TDM
// TDM: DMA a rows x 32 bf16 tile (global row stride 1024 elems) into LDS with
// padding: 64B rows + 16B pad -> 80B (40-elem) LDS row stride, bank-conflict-free.
__device__ __forceinline__ void tdm_load_2d(const u16* g, u16* l, unsigned rows) {
    const unsigned long long ga = (unsigned long long)(size_t)g;
    const unsigned la = (unsigned)(size_t)l;   // generic LDS addr: low 32 bits = LDS offset
    u32x4 g0;
    g0[0] = 1u;                                              // count=1 (valid, user mode)
    g0[1] = la;                                              // lds_addr
    g0[2] = (unsigned)ga;                                    // global_addr[31:0]
    g0[3] = ((unsigned)(ga >> 32) & 0x01FFFFFFu)             // global_addr[56:32]
            | 0x80000000u;                                   // type=2 ("image")
    i32x8 g1;
    g1[0] = 0x06D10000;            // data_size=2B | pad_enable | pad_interval=16DW | pad_amount=4DW
    g1[1] = (int)(1024u << 16);    // tensor_dim0 = 1024 (bits 79:48, low half)
    g1[2] = (int)(rows  << 16);    // tensor_dim0 hi=0 | tensor_dim1 = rows
    g1[3] = (int)(32u   << 16);    // tensor_dim1 hi=0 | tile_dim0 = 32
    g1[4] = (int)rows;             // tile_dim1 = rows | tile_dim2 = 0
    g1[5] = 1024;                  // tensor_dim0_stride[31:0] = 1024
    g1[6] = 0;                     // stride hi | tensor_dim1_stride lo
    g1[7] = 0;
    const i32x4 gz4 = {0, 0, 0, 0};                // groups 2/3: 2-D tile
    const i32x8 gz8 = {0, 0, 0, 0, 0, 0, 0, 0};    // extra group (clang-23 6-arg form)
    __builtin_amdgcn_tensor_load_to_lds(g0, g1, gz4, gz4, gz8, 0);
}
#endif

// ---------------------------------------------------------------- cvt f32 -> bf16
__global__ void cvt_f32_bf16(const float* __restrict__ s, u16* __restrict__ d, int n4) {
    int i = blockIdx.x * blockDim.x + threadIdx.x;
    if (i >= n4) return;
    float4 f = ((const float4*)s)[i];
    union { u16 u[4]; uint2 v; } r;
    r.u[0] = f2bf(f.x); r.u[1] = f2bf(f.y); r.u[2] = f2bf(f.z); r.u[3] = f2bf(f.w);
    ((uint2*)d)[i] = r.v;
}

// ---------------------------------------------------------------- QKV projection
// z = x @ W^T + b ; scatter into Q/K head-major [nb,h,C,64] and V transposed [nb,h,64,C]
// Block: 256 threads = 8 waves as 4(M) x 2(N); wave tile 64x32; block tile 256x64.
// A (256x32) and B (64x32) bf16 tiles staged in LDS by the Tensor Data Mover,
// double buffered; wave 0 issues TDM + waits TENSORcnt, block barrier publishes.
#define BM 256
#define BN 64
#define BK 32
#define LSTR 40   // padded LDS row stride in elements (80 bytes)

__global__ __launch_bounds__(256) void qkv_gemm(const u16* __restrict__ xb,
                                                const u16* __restrict__ Wb,
                                                const float* __restrict__ bias,
                                                u16* __restrict__ Qh,
                                                u16* __restrict__ Kh,
                                                u16* __restrict__ Vt) {
    __shared__ __align__(16) u16 As[2][BM * LSTR];   // 2 x 20KB
    __shared__ __align__(16) u16 Bs[2][BN * LSTR];   // 2 x 5KB

    const int tid  = threadIdx.x;
    const int lane = tid & 31;
    const int wid  = tid >> 5;
    const int l15  = lane & 15;
    const int half = lane >> 4;
    const int kb   = half << 3;

    const int wm = wid & 3;
    const int wn = wid >> 2;
    const int m0blk = blockIdx.x * BM;
    const int n0blk = blockIdx.y * BN;

    auto stage = [&](int buf, int k0) {
#if USE_TDM
        if (wid == 0) {
            tdm_load_2d(xb + (size_t)m0blk * D_MODEL + k0, &As[buf][0], BM);
            tdm_load_2d(Wb + (size_t)n0blk * D_MODEL + k0, &Bs[buf][0], BN);
        }
#else
        // cooperative fallback: A: row per thread, 4 x b128; B: 64 rows x 4 chunks
        const u16* ga = xb + (size_t)(m0blk + tid) * D_MODEL + k0;
        #pragma unroll
        for (int c = 0; c < 4; ++c)
            *(uint4*)&As[buf][tid * LSTR + c * 8] = *(const uint4*)(ga + c * 8);
        *(uint4*)&Bs[buf][(tid >> 2) * LSTR + (tid & 3) * 8] =
            *(const uint4*)(Wb + (size_t)(n0blk + (tid >> 2)) * D_MODEL + k0 + (tid & 3) * 8);
#endif
    };

    const v8f vzero = {0.f,0.f,0.f,0.f,0.f,0.f,0.f,0.f};
    v8f acc[4][2];
    #pragma unroll
    for (int i = 0; i < 4; ++i)
        #pragma unroll
        for (int j = 0; j < 2; ++j) acc[i][j] = vzero;

    stage(0, 0);

    for (int k0 = 0; k0 < D_MODEL; k0 += BK) {
        const int  buf  = (k0 >> 5) & 1;
        const bool more = (k0 + BK) < D_MODEL;
        if (more) stage(buf ^ 1, k0 + BK);
#if USE_TDM
        if (wid == 0) {
            if (more) { WAIT_TENSORCNT(2); }   // in-order: current tile's 2 ops done
            else      { WAIT_TENSORCNT(0); }
        }
#endif
        __syncthreads();   // publish current buffer to all waves

        v16bf af[4], bw[2];
        #pragma unroll
        for (int i = 0; i < 4; ++i)
            af[i] = load_frag(&As[buf][(wm * 64 + i * 16 + l15) * LSTR], kb);
        #pragma unroll
        for (int j = 0; j < 2; ++j)
            bw[j] = load_frag(&Bs[buf][(wn * 32 + j * 16 + l15) * LSTR], kb);
        #pragma unroll
        for (int i = 0; i < 4; ++i)
            #pragma unroll
            for (int j = 0; j < 2; ++j)
                acc[i][j] = wmma_bf16(af[i], bw[j], acc[i][j]);

        __syncthreads();   // all reads done before this buffer is re-staged
    }

    // Epilogue: bias + scatter. C/D layout: lane l15 = N, VGPR r -> M = half*8 + r.
    #pragma unroll
    for (int j = 0; j < 2; ++j) {
        const int nn  = n0blk + wn * 32 + j * 16 + l15;
        const float bv = bias[nn];
        const int sec = nn >> 10;          // 0=K, 1=Q, 2=V (reference order)
        const int d   = nn & 1023;
        const int h   = d >> 6;
        const int e   = d & 63;
        #pragma unroll
        for (int i = 0; i < 4; ++i) {
            #pragma unroll
            for (int r = 0; r < 8; ++r) {
                const int mm    = m0blk + wm * 64 + i * 16 + half * 8 + r;
                const int batch = mm >> 11;
                const int c     = mm & 2047;
                const u16 val   = f2bf(acc[i][j][r] + bv);
                const size_t bh = (size_t)(batch * N_HEADS + h);
                if (sec == 0)       Kh[((bh * SEQ + c) << 6) + e] = val;
                else if (sec == 1)  Qh[((bh * SEQ + c) << 6) + e] = val;
                else                Vt[((bh << 6) + e) * SEQ + c] = val;
            }
        }
    }
}

// ---------------------------------------------------------------- causal attention
// One wave per 16-query tile; flash-style streaming over 32-key chunks.
__global__ __launch_bounds__(128) void attn(const u16* __restrict__ Qh,
                                            const u16* __restrict__ Kh,
                                            const u16* __restrict__ Vt,
                                            float* __restrict__ out) {
    __shared__ __align__(16) u16 pb[4][16 * 32];   // per-wave P staging (C-layout -> A-layout)

    const int lane = threadIdx.x & 31;
    const int wid  = threadIdx.x >> 5;
    const int l15  = lane & 15;
    const int half = lane >> 4;
    const int kb   = half << 3;

    const int t  = blockIdx.x * 4 + wid;     // global 16-query tile id
    const int nb = t >> 11;                  // 16 heads * 128 tiles per batch
    const int h  = (t >> 7) & 15;
    const int q0 = (t & 127) << 4;

    const size_t base = (size_t)(nb * N_HEADS + h) << 17;   // 2048*64 elems per (nb,h)
    const u16* Qp = Qh + base;
    const u16* Kp = Kh + base;
    const u16* Vp = Vt + base;
    u16* myp = pb[wid];

    v16bf aq[2];
    #pragma unroll
    for (int kk = 0; kk < 2; ++kk)
        aq[kk] = load_frag(Qp + (size_t)(q0 + l15) * HEAD_DIM + kk * 32, kb);

    const v8f vzero = {0.f,0.f,0.f,0.f,0.f,0.f,0.f,0.f};
    v8f o[4] = {vzero, vzero, vzero, vzero};
    float mrow[8], lrow[8];
    #pragma unroll
    for (int r = 0; r < 8; ++r) { mrow[r] = -__builtin_inff(); lrow[r] = 0.f; }

    const int qhi = q0 + 15;
    for (int j0 = 0; j0 <= qhi; j0 += 32) {
        // ---- S = Q K^T for 32 keys (two 16-key subtiles), accumulated over hd=64
        v8f s0 = vzero, s1 = vzero;
        #pragma unroll
        for (int kk = 0; kk < 2; ++kk) {
            v16bf bk0 = load_frag(Kp + (size_t)(j0 + l15)      * HEAD_DIM + kk * 32, kb);
            v16bf bk1 = load_frag(Kp + (size_t)(j0 + 16 + l15) * HEAD_DIM + kk * 32, kb);
            s0 = wmma_bf16(aq[kk], bk0, s0);
            s1 = wmma_bf16(aq[kk], bk1, s1);
        }

        // ---- scale + causal mask (only chunks crossing the diagonal need it)
        const bool edge = (j0 + 31 > q0);
        #pragma unroll
        for (int r = 0; r < 8; ++r) {
            const int q = q0 + half * 8 + r;
            float a = s0[r] * SCALE;
            float c = s1[r] * SCALE;
            if (edge) {
                if (j0 + l15      > q) a = -__builtin_inff();
                if (j0 + 16 + l15 > q) c = -__builtin_inff();
            }
            s0[r] = a; s1[r] = c;
        }

        // ---- online softmax: row reductions across the 16-lane half (xor 1..8)
        float alpha[8];
        #pragma unroll
        for (int r = 0; r < 8; ++r) {
            float v = fmaxf(s0[r], s1[r]);
            #pragma unroll
            for (int off = 8; off >= 1; off >>= 1)
                v = fmaxf(v, __shfl_xor(v, off, 32));
            const float mn = fmaxf(mrow[r], v);
            alpha[r] = __expf(mrow[r] - mn);
            mrow[r]  = mn;
            s0[r] = __expf(s0[r] - mn);
            s1[r] = __expf(s1[r] - mn);
            float rs = s0[r] + s1[r];
            #pragma unroll
            for (int off = 8; off >= 1; off >>= 1)
                rs += __shfl_xor(rs, off, 32);
            lrow[r] = lrow[r] * alpha[r] + rs;
        }
        #pragma unroll
        for (int tt = 0; tt < 4; ++tt)
            #pragma unroll
            for (int r = 0; r < 8; ++r)
                o[tt][r] *= alpha[r];

        // ---- P (16x32) C-layout -> LDS -> A-layout fragment
        #pragma unroll
        for (int r = 0; r < 8; ++r) {
            const int row = half * 8 + r;
            myp[row * 32 + l15]      = f2bf(s0[r]);
            myp[row * 32 + 16 + l15] = f2bf(s1[r]);
        }
        asm volatile("s_wait_dscnt 0" ::: "memory");
        v16bf ap = load_frag(myp + l15 * 32, kb);

        // ---- O += P V  (V transposed: lane-contiguous B fragments)
        #pragma unroll
        for (int dsub = 0; dsub < 4; ++dsub) {
            v16bf bv = load_frag(Vp + (size_t)(dsub * 16 + l15) * SEQ + j0, kb);
            o[dsub] = wmma_bf16(ap, bv, o[dsub]);
        }
    }

    // ---- normalize and store: out[nb, c, h*64 + d]
    #pragma unroll
    for (int dsub = 0; dsub < 4; ++dsub) {
        #pragma unroll
        for (int r = 0; r < 8; ++r) {
            const int c = q0 + half * 8 + r;
            out[((size_t)(nb * SEQ + c) << 10) + h * 64 + dsub * 16 + l15] = o[dsub][r] / lrow[r];
        }
    }
}

// ---------------------------------------------------------------- launcher
extern "C" void kernel_launch(void* const* d_in, const int* in_sizes, int n_in,
                              void* d_out, int out_size, void* d_ws, size_t ws_size,
                              hipStream_t stream) {
    const float* x = (const float*)d_in[0];   // [4,2048,1024]
    const float* W = (const float*)d_in[1];   // [3072,1024]
    const float* b = (const float*)d_in[2];   // [3072]
    float* out = (float*)d_out;               // [4,2048,1024]

    const size_t QE = (size_t)N_BATCH * N_HEADS * SEQ * HEAD_DIM;   // 8,388,608
    u16* Qh = (u16*)d_ws;
    u16* Kh = Qh + QE;
    u16* Vt = Kh + QE;
    u16* xb = Vt + QE;
    u16* Wb = xb + (size_t)N_BATCH * SEQ * D_MODEL;

    const int xn4 = N_BATCH * SEQ * D_MODEL / 4;      // 2,097,152
    const int wn4 = 3 * D_MODEL * D_MODEL / 4;        //   786,432
    cvt_f32_bf16<<<(xn4 + 255) / 256, 256, 0, stream>>>(x, xb, xn4);
    cvt_f32_bf16<<<(wn4 + 255) / 256, 256, 0, stream>>>(W, Wb, wn4);

    dim3 gg(8192 / BM, 3072 / BN);                    // (32, 48)
    qkv_gemm<<<gg, 256, 0, stream>>>(xb, Wb, b, Qh, Kh, Vt);

    const int tiles = N_BATCH * N_HEADS * (SEQ / 16); // 8192 wave tiles
    attn<<<tiles / 4, 128, 0, stream>>>(Qh, Kh, Vt, out);
}
